// GroupedMLP_12455405158654
// MI455X (gfx1250) — matrix operand order repaired and checked by reading, not compiled, not run on hardware
//
#include <hip/hip_runtime.h>
#include <hip/hip_bf16.h>

// ---- vector types matching the gfx1250 WMMA builtin signatures ----
typedef __attribute__((ext_vector_type(16))) _Float16 v16h;
typedef __attribute__((ext_vector_type(8)))  _Float16 v8h;
typedef __attribute__((ext_vector_type(4)))  _Float16 v4h;
typedef __attribute__((ext_vector_type(8)))  float    v8f;
typedef __attribute__((ext_vector_type(4)))  float    v4f;

// Problem constants (fixed by the reference)
constexpr int TOK   = 4096;
constexpr int HID   = 1024;
constexpr int INTR  = 2048;
constexpr int NEXP  = 8;

// Tiling
constexpr int BM  = 128;   // token rows per block (expert boundaries are 128-aligned)
constexpr int BN  = 64;    // output cols per block
constexpr int BK  = 64;    // K per LDS round = 2 WMMA K-steps (halves barrier rounds)
constexpr int WK  = 32;    // WMMA K
constexpr int LDA = 72;    // padded LDS row stride in halves: 144B rows, 16B-aligned chunks

// Expert id of a token row: # of cumulative-count prefixes <= row.
__device__ __forceinline__ int expert_of(const int* __restrict__ tpe, int row) {
    int acc = 0, g = 0;
#pragma unroll
    for (int i = 0; i < NEXP; ++i) { acc += tpe[i]; g += (row >= acc) ? 1 : 0; }
    return g;
}

__device__ __forceinline__ v16h make_frag(v8h lo, v8h hi) {
    union { v16h v; v8h h[2]; } u;
    u.h[0] = lo; u.h[1] = hi;
    return u.v;
}

// =====================================================================
// Kernel 1: h = x @ w1[g]  (up | gate), act = silu(up) * gate  -> f16 ws
// =====================================================================
__global__ __launch_bounds__(256)
void gmlp_fc1_swiglu(const float* __restrict__ x,     // [TOK][HID] f32
                     const int*   __restrict__ tpe,   // [NEXP]
                     const float* __restrict__ w1,    // [NEXP][HID][2*INTR] f32
                     _Float16*    __restrict__ act)   // [TOK][INTR] f16 (workspace)
{
    __shared__ _Float16 As [BM][LDA];   // token tile, f16                (18.4 KB)
    __shared__ _Float16 BuS[BN][LDA];   // up-weight tile, transposed [n][k]  (9.2 KB)
    __shared__ _Float16 BgS[BN][LDA];   // gate-weight tile, transposed [n][k](9.2 KB)

    const int tid    = threadIdx.x;
    const int lane   = tid & 31;
    const int wave   = tid >> 5;
    const int laneLo = lane & 15;
    const int laneHi = lane >> 4;
    const int m0     = blockIdx.y * BM;
    const int n0     = blockIdx.x * BN;
    const int wm     = (wave >> 1) * 32;   // 4 waves tile M
    const int wn     = (wave & 1) * 32;    // 2 waves tile N

    const int g = expert_of(tpe, m0);
    const float* wg = w1 + (size_t)g * HID * (2 * INTR);

    v8f accU[2][2] = {};
    v8f accG[2][2] = {};

    const int ar = tid >> 4, ac = tid & 15;   // A: 16 rows x 16 float4 per pass, 8 passes
    const int br = tid >> 4, bc = tid & 15;   // B: 16 K-rows x 16 float4 per pass, 4 passes

    for (int k0 = 0; k0 < HID; k0 += BK) {
        // ---- token tile: fp32 -> f16 into LDS (row-major [m][k]) ----
#pragma unroll
        for (int p = 0; p < 8; ++p) {
            const int row = ar + p * 16;
            v4f a = *(const v4f*)(x + (size_t)(m0 + row) * HID + k0 + ac * 4);
            v4h h = { (_Float16)a.x, (_Float16)a.y, (_Float16)a.z, (_Float16)a.w };
            *(v4h*)&As[row][ac * 4] = h;
        }
        // ---- weight tiles (up @ n0, gate @ n0+INTR): fp32 -> f16, transposed [n][k] ----
#pragma unroll
        for (int p = 0; p < 4; ++p) {
            const int row = br + p * 16;                                   // K row
            const float* src = wg + (size_t)(k0 + row) * (2 * INTR) + n0 + bc * 4;
            v4f bu = *(const v4f*)(src);
            v4f bg = *(const v4f*)(src + INTR);
#pragma unroll
            for (int e = 0; e < 4; ++e) {
                BuS[bc * 4 + e][row] = (_Float16)bu[e];
                BgS[bc * 4 + e][row] = (_Float16)bg[e];
            }
            if (k0 + BK < HID) {                    // gfx1250 global_prefetch_b8
                __builtin_prefetch(src + (size_t)BK * (2 * INTR), 0, 1);
                __builtin_prefetch(src + (size_t)BK * (2 * INTR) + INTR, 0, 1);
            }
        }
        __syncthreads();

        // ---- two WMMA K-steps per LDS round ----
#pragma unroll
        for (int s = 0; s < 2; ++s) {
            const int ks = s * WK;
            v16h aF[2], buF[2], bgF[2];
#pragma unroll
            for (int i = 0; i < 2; ++i) {
                const int m = wm + 16 * i + laneLo;
                const int o = ks + laneHi * 16;        // A: K {o..o+15} contiguous
                aF[i] = make_frag(*(const v8h*)&As[m][o], *(const v8h*)&As[m][o + 8]);
            }
#pragma unroll
            for (int j = 0; j < 2; ++j) {
                const int n = wn + 16 * j + laneLo;
                const int o = ks + laneHi * 16;        // B: K {o..o+15} contiguous
                buF[j] = make_frag(*(const v8h*)&BuS[n][o], *(const v8h*)&BuS[n][o + 8]);
                bgF[j] = make_frag(*(const v8h*)&BgS[n][o], *(const v8h*)&BgS[n][o + 8]);
            }
#pragma unroll
            for (int i = 0; i < 2; ++i)
#pragma unroll
                for (int j = 0; j < 2; ++j) {
                    accU[i][j] = __builtin_amdgcn_wmma_f32_16x16x32_f16(
                        false, aF[i], false, buF[j], (short)0, accU[i][j], false, false);
                    accG[i][j] = __builtin_amdgcn_wmma_f32_16x16x32_f16(
                        false, aF[i], false, bgF[j], (short)0, accG[i][j], false, false);
                }
        }
        __syncthreads();
    }

    // ---- SwiGLU in registers, store activation tile as f16 ----
#pragma unroll
    for (int i = 0; i < 2; ++i)
#pragma unroll
        for (int j = 0; j < 2; ++j) {
            const int n = n0 + wn + 16 * j + laneLo;
#pragma unroll
            for (int r = 0; r < 8; ++r) {
                const int m = m0 + wm + 16 * i + laneHi * 8 + r;   // C/D layout: VGPR r -> M
                const float up = accU[i][j][r];
                const float gt = accG[i][j][r];
                const float s  = up / (1.0f + __expf(-up));        // silu
                act[(size_t)m * INTR + n] = (_Float16)(s * gt);
            }
        }
}

// =====================================================================
// Kernel 2: out = act(f16) @ w2[g]  -> f32
// =====================================================================
__global__ __launch_bounds__(256)
void gmlp_fc2(const _Float16* __restrict__ act,   // [TOK][INTR] f16
              const int*      __restrict__ tpe,   // [NEXP]
              const float*    __restrict__ w2,    // [NEXP][INTR][HID] f32
              float*          __restrict__ out)   // [TOK][HID] f32
{
    __shared__ _Float16 As[BM][LDA];   // 18.4 KB
    __shared__ _Float16 Bs[BN][LDA];   //  9.2 KB

    const int tid    = threadIdx.x;
    const int lane   = tid & 31;
    const int wave   = tid >> 5;
    const int laneLo = lane & 15;
    const int laneHi = lane >> 4;
    const int m0     = blockIdx.y * BM;
    const int n0     = blockIdx.x * BN;
    const int wm     = (wave >> 1) * 32;
    const int wn     = (wave & 1) * 32;

    const int g = expert_of(tpe, m0);
    const float* wg = w2 + (size_t)g * INTR * HID;

    v8f acc[2][2] = {};

    const int ar = tid >> 3, ac = tid & 7;    // A: 32 rows x 8 v8h per pass, 4 passes
    const int br = tid >> 4, bc = tid & 15;   // B: 16 K-rows x 16 float4 per pass, 4 passes

    for (int k0 = 0; k0 < INTR; k0 += BK) {
        // ---- activation tile: already f16, straight b128 copy into LDS ----
#pragma unroll
        for (int p = 0; p < 4; ++p) {
            const int row = ar + p * 32;
            v8h h = *(const v8h*)(act + (size_t)(m0 + row) * INTR + k0 + ac * 8);
            *(v8h*)&As[row][ac * 8] = h;
        }
        // ---- w2 tile: fp32 -> f16, transposed [n][k] ----
#pragma unroll
        for (int p = 0; p < 4; ++p) {
            const int row = br + p * 16;
            const float* src = wg + (size_t)(k0 + row) * HID + n0 + bc * 4;
            v4f b = *(const v4f*)(src);
#pragma unroll
            for (int e = 0; e < 4; ++e)
                Bs[bc * 4 + e][row] = (_Float16)b[e];
            if (k0 + BK < INTR)
                __builtin_prefetch(src + (size_t)BK * HID, 0, 1);
        }
        __syncthreads();

#pragma unroll
        for (int s = 0; s < 2; ++s) {
            const int ks = s * WK;
            v16h aF[2], bF[2];
#pragma unroll
            for (int i = 0; i < 2; ++i) {
                const int m = wm + 16 * i + laneLo;
                const int o = ks + laneHi * 16;
                aF[i] = make_frag(*(const v8h*)&As[m][o], *(const v8h*)&As[m][o + 8]);
            }
#pragma unroll
            for (int j = 0; j < 2; ++j) {
                const int n = wn + 16 * j + laneLo;
                const int o = ks + laneHi * 16;
                bF[j] = make_frag(*(const v8h*)&Bs[n][o], *(const v8h*)&Bs[n][o + 8]);
            }
#pragma unroll
            for (int i = 0; i < 2; ++i)
#pragma unroll
                for (int j = 0; j < 2; ++j)
                    acc[i][j] = __builtin_amdgcn_wmma_f32_16x16x32_f16(
                        false, aF[i], false, bF[j], (short)0, acc[i][j], false, false);
        }
        __syncthreads();
    }

#pragma unroll
    for (int i = 0; i < 2; ++i)
#pragma unroll
        for (int j = 0; j < 2; ++j) {
            const int n = n0 + wn + 16 * j + laneLo;
#pragma unroll
            for (int r = 0; r < 8; ++r) {
                const int m = m0 + wm + 16 * i + laneHi * 8 + r;
                out[(size_t)m * HID + n] = acc[i][j][r];
            }
        }
}

extern "C" void kernel_launch(void* const* d_in, const int* in_sizes, int n_in,
                              void* d_out, int out_size, void* d_ws, size_t ws_size,
                              hipStream_t stream) {
    const float* x   = (const float*)d_in[0];   // permuted_tokens [4096,1024]
    const int*   tpe = (const int*)  d_in[1];   // tokens_per_expert [8]
    const float* w1  = (const float*)d_in[2];   // [8,1024,4096]
    const float* w2  = (const float*)d_in[3];   // [8,2048,1024]
    float*     outp = (float*)d_out;            // [4096,1024]
    _Float16*  actb = (_Float16*)d_ws;          // [4096,2048] f16 = 16 MB scratch

    (void)in_sizes; (void)n_in; (void)out_size; (void)ws_size;

    dim3 blk(256);
    gmlp_fc1_swiglu<<<dim3(INTR / BN, TOK / BM), blk, 0, stream>>>(x, tpe, w1, actb);
    gmlp_fc2      <<<dim3(HID  / BN, TOK / BM), blk, 0, stream>>>(actb, tpe, w2, outp);
}
